// Transformer_39857296507469
// MI455X (gfx1250) — matrix operand order repaired and checked by reading, not compiled, not run on hardware
//
#include <hip/hip_runtime.h>

// ---------------- static config ----------------
#define M_TOTAL 6400          // B*N rows
#define N_SEQ   3200
#define DM      256
#define HD      32
#define SUP     32
#define DFF     1024
#define ENC_D   26

#define QK_SCALE 0.42044820762685725f   // 32^-0.25
#define RS_M     0.17677669529663687f   // 1/sqrt(32)
#define LN_EPS   1e-6f

typedef __attribute__((ext_vector_type(16))) __bf16 v16bf;
typedef __attribute__((ext_vector_type(8)))  __bf16 v8bf;
typedef __attribute__((ext_vector_type(8)))  float  v8f;

__device__ __forceinline__ float wred_sum(float x) {
  #pragma unroll
  for (int o = 16; o; o >>= 1) x += __shfl_xor(x, o, 32);
  return x;
}
__device__ __forceinline__ float wred_max(float x) {
  #pragma unroll
  for (int o = 16; o; o >>= 1) x = fmaxf(x, __shfl_xor(x, o, 32));
  return x;
}
__device__ __forceinline__ unsigned fenc(float f) {
  unsigned u = __float_as_uint(f);
  return (u & 0x80000000u) ? ~u : (u | 0x80000000u);
}
__device__ __forceinline__ float fdec(unsigned e) {
  unsigned u = (e & 0x80000000u) ? (e & 0x7FFFFFFFu) : ~e;
  return __uint_as_float(u);
}
__device__ __forceinline__ v16bf ldb16(const __bf16* __restrict__ p) {
  const v8bf* q = (const v8bf*)p;          // 32B-aligned by construction
  return __builtin_shufflevector(q[0], q[1],
         0,1,2,3,4,5,6,7,8,9,10,11,12,13,14,15);
}

// -------- weight prep: fp32 [K,N] -> bf16 transposed zero-padded [NPAD, KPAD] --------
__global__ void k_wprep(const float* __restrict__ W, __bf16* __restrict__ Wt,
                        int K, int Nout, int KPAD, int NPAD) {
  int i = blockIdx.x * blockDim.x + threadIdx.x;
  if (i >= KPAD * NPAD) return;
  int n = i / KPAD, k = i % KPAD;
  float v = (k < K && n < Nout) ? W[(size_t)k * Nout + n] : 0.f;
  Wt[(size_t)n * KPAD + k] = (__bf16)v;
}

// ---------------- WMMA GEMM ----------------
// A fp32 [M, KPAD] (zero-padded), Wt bf16 [NPAD, KPAD] transposed+padded.
// One wave computes a 16x64 C tile. Per k-step: 4 A b128 loads + 8 B b128 loads
// batched up-front, then 4 back-to-back WMMAs on distinct accumulators.
// No lane-divergent control flow before/around the WMMA loop (EXEC all ones).
__global__ __launch_bounds__(256)
void gemm_wmma(const float* __restrict__ A, const __bf16* __restrict__ Wt,
               const float* __restrict__ bias, const float* __restrict__ res,
               float* __restrict__ C, int Mrows, int KPAD, int Nout, int NPAD,
               int act) {
  int gwave = (blockIdx.x * blockDim.x + threadIdx.x) >> 5;
  int lane  = threadIdx.x & 31;
  int mtiles = Mrows >> 4;
  int ntiles = NPAD >> 6;
  if (gwave >= mtiles * ntiles) return;            // wave-uniform
  int mt = gwave / ntiles, nt = gwave % ntiles;
  int row0 = mt << 4, col0 = nt << 6;
  int l15 = lane & 15;
  int kgA = (lane < 16) ? 0 : 16;                  // A K contiguous per half-wave (matches B)
  int kgB = (lane < 16) ? 0 : 16;                  // B K contiguous per half-wave (matches A)

  const float*  Arow = A  + (size_t)(row0 + l15) * KPAD;
  const __bf16* Wc   = Wt + (size_t)(col0 + l15) * KPAD;
  const size_t  cstr = (size_t)16 * KPAD;          // 16 columns of Wt

  v8f acc[4] = {v8f{}, v8f{}, v8f{}, v8f{}};
  #pragma unroll 2
  for (int k0 = 0; k0 < KPAD; k0 += 32) {
    // batch all loads for this k-step
    const float4* ap = (const float4*)(Arow + k0 + kgA);
    float4 x0 = ap[0], x1 = ap[1], x2 = ap[2], x3 = ap[3];
    const __bf16* bp = Wc + k0 + kgB;
    v16bf b0 = ldb16(bp);
    v16bf b1 = ldb16(bp + cstr);
    v16bf b2 = ldb16(bp + 2 * cstr);
    v16bf b3 = ldb16(bp + 3 * cstr);
    v16bf a;
    a[0]=(__bf16)x0.x;  a[1]=(__bf16)x0.y;  a[2]=(__bf16)x0.z;  a[3]=(__bf16)x0.w;
    a[4]=(__bf16)x1.x;  a[5]=(__bf16)x1.y;  a[6]=(__bf16)x1.z;  a[7]=(__bf16)x1.w;
    a[8]=(__bf16)x2.x;  a[9]=(__bf16)x2.y;  a[10]=(__bf16)x2.z; a[11]=(__bf16)x2.w;
    a[12]=(__bf16)x3.x; a[13]=(__bf16)x3.y; a[14]=(__bf16)x3.z; a[15]=(__bf16)x3.w;
    // dependency-free WMMA burst (distinct accumulators)
    acc[0] = __builtin_amdgcn_wmma_f32_16x16x32_bf16(false, a, false, b0,
                                                     (short)0, acc[0], false, false);
    acc[1] = __builtin_amdgcn_wmma_f32_16x16x32_bf16(false, a, false, b1,
                                                     (short)0, acc[1], false, false);
    acc[2] = __builtin_amdgcn_wmma_f32_16x16x32_bf16(false, a, false, b2,
                                                     (short)0, acc[2], false, false);
    acc[3] = __builtin_amdgcn_wmma_f32_16x16x32_bf16(false, a, false, b3,
                                                     (short)0, acc[3], false, false);
  }
  // D layout: elem r -> row = row0 + r + (lane<16?0:8), col
  int rb = (lane < 16) ? 0 : 8;
  #pragma unroll
  for (int t = 0; t < 4; ++t) {
    int bcol = col0 + t * 16 + l15;
    if (bcol >= Nout) continue;                    // divergence only in epilogue
    float bv = bias ? bias[bcol] : 0.f;
    #pragma unroll
    for (int r = 0; r < 8; ++r) {
      int row = row0 + rb + r;
      float v = acc[t][r] + bv;
      if (act == 1) v = (v > 0.f) ? v : (__expf(v) - 1.f);   // ELU
      size_t idx = (size_t)row * Nout + bcol;
      if (res) v += res[idx];
      C[idx] = v;
    }
  }
}

// ---------------- LayerNorm: one wave per row, optional padded output width ----------------
__global__ __launch_bounds__(256)
void k_ln(const float* __restrict__ X, const float* __restrict__ g,
          const float* __restrict__ bt, float* __restrict__ Y, int D, int Dpad) {
  int wave = (blockIdx.x * blockDim.x + threadIdx.x) >> 5;
  int lane = threadIdx.x & 31;
  if (wave >= M_TOTAL) return;
  const float* xr = X + (size_t)wave * D;
  float vals[8];
  int cnt = (Dpad + 31) >> 5;
  float s = 0.f;
  for (int i = 0; i < cnt; ++i) {
    int c = lane + (i << 5);
    vals[i] = (c < D) ? xr[c] : 0.f;
    s += vals[i];
  }
  float mu = wred_sum(s) / (float)D;
  float v2 = 0.f;
  for (int i = 0; i < cnt; ++i) {
    int c = lane + (i << 5);
    float d = (c < D) ? (vals[i] - mu) : 0.f;
    v2 += d * d;
  }
  float rstd = rsqrtf(wred_sum(v2) / (float)D + LN_EPS);
  for (int i = 0; i < cnt; ++i) {
    int c = lane + (i << 5);
    if (c < Dpad)
      Y[(size_t)wave * Dpad + c] = (c < D) ? (vals[i] - mu) * rstd * g[c] + bt[c] : 0.f;
  }
}

// ---------------- one-hot + feature concat ----------------
__global__ void k_enc_build(const float* __restrict__ X, float* __restrict__ E) {
  int i = blockIdx.x * blockDim.x + threadIdx.x;
  if (i >= M_TOTAL * ENC_D) return;
  int row = i / ENC_D, c = i % ENC_D;
  float v;
  if (c < 12) { int id = (int)X[row * 15]; v = (c == id) ? 1.f : 0.f; }
  else        { v = X[row * 15 + 1 + (c - 12)]; }
  E[i] = v;
}

// ---------------- FAVOR+ phi. mode 0: query, mode 1: key pass-1 ----------------
__global__ __launch_bounds__(256)
void k_phi(const float* __restrict__ IN, const float* __restrict__ proj,
           float* __restrict__ OUT, unsigned* __restrict__ stab, int mode) {
  __shared__ float Wp[SUP * HD];
  __shared__ float xs[8][HD];
  int tid = threadIdx.x, wave = tid >> 5, lane = tid & 31;
  for (int i = tid; i < HD * SUP; i += 256) Wp[i] = proj[i];
  int job = blockIdx.x * 8 + wave;     // job = row*8 + h, exact grid
  int row = job >> 3, h = job & 7;
  float xv = IN[(size_t)row * DM + h * HD + lane] * QK_SCALE;
  xs[wave][lane] = xv;
  __syncthreads();
  float pr = 0.f, ss = 0.f;
  #pragma unroll
  for (int d = 0; d < HD; ++d) {
    float xd = xs[wave][d];
    pr += xd * Wp[d * SUP + lane];
    ss += xd * xd;
  }
  float hv = pr - 0.5f * ss;
  if (mode == 0) {
    float mx = wred_max(hv);
    OUT[(size_t)row * DM + h * HD + lane] = __expf(hv - mx) * RS_M;
  } else {
    OUT[(size_t)row * DM + h * HD + lane] = hv;
    float mx = wred_max(hv);
    if (lane == 0) {
      int bh = ((row >= N_SEQ) ? 8 : 0) + h;
      atomicMax(&stab[bh], fenc(mx));
    }
  }
}

__global__ void k_stab_init(unsigned* __restrict__ stab) {
  if (threadIdx.x < 16) stab[threadIdx.x] = 0u;
}

__global__ void k_kp(const float* __restrict__ KH, const unsigned* __restrict__ stab,
                     float* __restrict__ KP) {
  int i = blockIdx.x * blockDim.x + threadIdx.x;
  if (i >= M_TOTAL * DM) return;
  int row = i >> 8, col = i & 255;
  int bh = ((row >= N_SEQ) ? 8 : 0) + (col >> 5);
  KP[i] = __expf(KH[i] - fdec(stab[bh])) * RS_M;
}

// ---------------- kv[b,h,m,d] = sum_n kp[n,m]*v[n,d];  ksum[b,h,m] = sum_n kp[n,m] ----------------
__global__ __launch_bounds__(256)
void k_kvsum(const float* __restrict__ KP, const float* __restrict__ V,
             float* __restrict__ KV, float* __restrict__ KSUM) {
  int bh = blockIdx.x, b = bh >> 3, h = bh & 7;
  int tid = threadIdx.x;
  int m = tid >> 3, d0 = tid & 7;
  __shared__ float kpch[16][33];
  __shared__ float vch[16][33];
  float acc[4] = {0.f, 0.f, 0.f, 0.f};
  float ks = 0.f;
  size_t base = (size_t)b * N_SEQ * DM + (size_t)h * HD;
  for (int n0 = 0; n0 < N_SEQ; n0 += 16) {
    for (int i = tid; i < 16 * 32; i += 256) {
      int rr = i >> 5, cc = i & 31;
      size_t off = base + (size_t)(n0 + rr) * DM + cc;
      kpch[rr][cc] = KP[off];
      vch[rr][cc]  = V[off];
    }
    __syncthreads();
    #pragma unroll 4
    for (int rr = 0; rr < 16; ++rr) {
      float kpv = kpch[rr][m];
      if (d0 == 0) ks += kpv;
      #pragma unroll
      for (int i = 0; i < 4; ++i) acc[i] += kpv * vch[rr][d0 + 8 * i];
    }
    __syncthreads();
  }
  #pragma unroll
  for (int i = 0; i < 4; ++i) KV[bh * 1024 + m * 32 + d0 + 8 * i] = acc[i];
  if (d0 == 0) KSUM[bh * 32 + m] = ks;
}

// ---------------- out = (qp . kv) / (qp . ksum + eps) ----------------
__global__ __launch_bounds__(256)
void k_combine(const float* __restrict__ QP, const float* __restrict__ KV,
               const float* __restrict__ KSUM, float* __restrict__ OUT) {
  int tid = threadIdx.x, wave = tid >> 5, lane = tid & 31;
  int job = blockIdx.x * 8 + wave;
  int row = job >> 3, h = job & 7;
  int bh = ((row >= N_SEQ) ? 8 : 0) + h;
  float qpv = QP[(size_t)row * DM + h * HD + lane];
  float z = wred_sum(qpv * KSUM[bh * 32 + lane]) + 1e-6f;
  float acc = 0.f;
  #pragma unroll
  for (int m = 0; m < 32; ++m) {
    float qm = __shfl(qpv, m, 32);
    acc += qm * KV[bh * 1024 + m * 32 + lane];
  }
  OUT[(size_t)row * DM + h * HD + lane] = acc / z;
}

// ---------------- concats (padded widths, zero-filled) & finalize ----------------
__global__ void k_cat2(const float* __restrict__ A, const float* __restrict__ B2,
                       float* __restrict__ C, int da, int db, int dtpad) {
  int i = blockIdx.x * blockDim.x + threadIdx.x;
  if (i >= M_TOTAL * dtpad) return;
  int row = i / dtpad, c = i % dtpad;
  float v = 0.f;
  if (c < da)           v = A[(size_t)row * da + c];
  else if (c < da + db) v = B2[(size_t)row * db + (c - da)];
  C[i] = v;
}
__global__ void k_cat3(const float* __restrict__ A, const float* __restrict__ B2,
                       const float* __restrict__ C3, float* __restrict__ C,
                       int da, int db, int dc, int dtpad) {
  int i = blockIdx.x * blockDim.x + threadIdx.x;
  if (i >= M_TOTAL * dtpad) return;
  int row = i / dtpad, c = i % dtpad;
  float v = 0.f;
  if (c < da)                v = A[(size_t)row * da + c];
  else if (c < da + db)      v = B2[(size_t)row * db + (c - da)];
  else if (c < da + db + dc) v = C3[(size_t)row * dc + (c - da - db)];
  C[i] = v;
}
__global__ void k_final(const float* __restrict__ X, const float* __restrict__ lg,
                        const float* __restrict__ ch, const float* __restrict__ mo,
                        float* __restrict__ out) {
  int i = blockIdx.x * blockDim.x + threadIdx.x;
  if (i >= M_TOTAL * 12) return;
  int row = i / 12, c = i % 12;
  float msk = (X[row * 15] != 0.f) ? 1.f : 0.f;
  float v;
  if (c < 8)       v = lg[row * 8 + c];
  else if (c == 8) v = ch[row] * msk;
  else             v = mo[row * 3 + (c - 9)] * msk;
  out[i] = v;
}

// ================= host orchestration =================
struct DenseP { const float* b; const float* w; };
struct FfnP   { DenseP d1, d2; };
struct LnP    { const float* b; const float* g; };
struct MhaP   { const float* proj; DenseP wk, wo, wq, wv; };
struct EncP   { FfnP ffn; LnP ln1, ln2; MhaP mha; };
struct DecP   { FfnP ffn; LnP ln1, ln2, ln3; MhaP mha1, mha2; };

struct Cursor {
  void* const* in; int i;
  const float* next() { return (const float*)in[i++]; }
};
static DenseP rdDense(Cursor& c){ DenseP d; d.b=c.next(); d.w=c.next(); return d; }
static FfnP   rdFfn  (Cursor& c){ FfnP f; f.d1=rdDense(c); f.d2=rdDense(c); return f; }
static LnP    rdLn   (Cursor& c){ LnP l; l.b=c.next(); l.g=c.next(); return l; }
static MhaP   rdMha  (Cursor& c){ MhaP m; m.proj=c.next(); m.wk=rdDense(c);
                                  m.wo=rdDense(c); m.wq=rdDense(c); m.wv=rdDense(c); return m; }
static EncP   rdEnc  (Cursor& c){ EncP e; e.ffn=rdFfn(c); e.ln1=rdLn(c); e.ln2=rdLn(c);
                                  e.mha=rdMha(c); return e; }
static DecP   rdDec  (Cursor& c){ DecP d; d.ffn=rdFfn(c); d.ln1=rdLn(c); d.ln2=rdLn(c);
                                  d.ln3=rdLn(c); d.mha1=rdMha(c); d.mha2=rdMha(c); return d; }

// prepared (transposed, padded, bf16) dense
struct GW { const __bf16* wt; const float* bias; int kpad; int npad; };
struct MhaG { const float* proj; GW wq, wk, wv, wo; };
struct EncG { GW d1, d2; LnP ln1, ln2; MhaG mha; };
struct DecG { GW d1, d2; LnP ln1, ln2, ln3; MhaG mha1, mha2; };

struct Bufs {
  float *enc26, *enc26n, *enc_t, *eo, *dcur, *t1, *t2;
  float *q, *k, *v, *qp, *kh, *attno, *hid;
  float *cat282, *cat290, *logits8, *charge, *mom, *kv, *ksum;
  unsigned* stab;
};

static inline int cdiv(long a, long b) { return (int)((a + b - 1) / b); }

static GW prep_dense(hipStream_t s, const DenseP& d, int K, int N, __bf16*& pool) {
  int kpad = (K + 31) & ~31;
  int npad = (N + 63) & ~63;
  GW g; g.wt = pool; g.bias = d.b; g.kpad = kpad; g.npad = npad;
  long tot = (long)kpad * npad;
  k_wprep<<<cdiv(tot, 256), 256, 0, s>>>(d.w, pool, K, N, kpad, npad);
  pool += tot;
  return g;
}
static MhaG prep_mha(hipStream_t s, const MhaP& m, __bf16*& pool) {
  MhaG g; g.proj = m.proj;
  g.wq = prep_dense(s, m.wq, DM, DM, pool);
  g.wk = prep_dense(s, m.wk, DM, DM, pool);
  g.wv = prep_dense(s, m.wv, DM, DM, pool);
  g.wo = prep_dense(s, m.wo, DM, DM, pool);
  return g;
}
static EncG prep_enc(hipStream_t s, const EncP& e, __bf16*& pool) {
  EncG g;
  g.d1 = prep_dense(s, e.ffn.d1, DM, DFF, pool);
  g.d2 = prep_dense(s, e.ffn.d2, DFF, DM, pool);
  g.ln1 = e.ln1; g.ln2 = e.ln2;
  g.mha = prep_mha(s, e.mha, pool);
  return g;
}
static DecG prep_dec(hipStream_t s, const DecP& d, __bf16*& pool) {
  DecG g;
  g.d1 = prep_dense(s, d.ffn.d1, DM, DFF, pool);
  g.d2 = prep_dense(s, d.ffn.d2, DFF, DM, pool);
  g.ln1 = d.ln1; g.ln2 = d.ln2; g.ln3 = d.ln3;
  g.mha1 = prep_mha(s, d.mha1, pool);
  g.mha2 = prep_mha(s, d.mha2, pool);
  return g;
}

static void run_gemm(hipStream_t s, const float* A, const GW& g, const float* res,
                     float* C, int Nout, int act) {
  long waves = (long)(M_TOTAL >> 4) * (g.npad >> 6);
  gemm_wmma<<<cdiv(waves * 32, 256), 256, 0, s>>>(A, g.wt, g.bias, res, C,
                                                  M_TOTAL, g.kpad, Nout, g.npad, act);
}
static void run_ln(hipStream_t s, const float* X, const LnP& p, float* Y, int D, int Dpad) {
  k_ln<<<cdiv((long)M_TOTAL * 32, 256), 256, 0, s>>>(X, p.g, p.b, Y, D, Dpad);
}
static void run_performer(hipStream_t s, const MhaG& p, const float* qin, const float* kin,
                          const float* vin, const float* resid, float* dst, Bufs& w) {
  run_gemm(s, qin, p.wq, nullptr, w.q, DM, 0);
  run_gemm(s, kin, p.wk, nullptr, w.k, DM, 0);
  run_gemm(s, vin, p.wv, nullptr, w.v, DM, 0);
  k_phi<<<M_TOTAL, 256, 0, s>>>(w.q, p.proj, w.qp, w.stab, 0);
  k_stab_init<<<1, 64, 0, s>>>(w.stab);
  k_phi<<<M_TOTAL, 256, 0, s>>>(w.k, p.proj, w.kh, w.stab, 1);
  k_kp<<<cdiv((long)M_TOTAL * DM, 256), 256, 0, s>>>(w.kh, w.stab, w.kh);
  k_kvsum<<<16, 256, 0, s>>>(w.kh, w.v, w.kv, w.ksum);
  k_combine<<<M_TOTAL, 256, 0, s>>>(w.qp, w.kv, w.ksum, w.attno);
  run_gemm(s, w.attno, p.wo, resid, dst, DM, 0);
}
static void run_enc(hipStream_t s, const EncG& p, const float* xin, float* xout, Bufs& w) {
  run_performer(s, p.mha, xin, xin, xin, xin, w.t2, w);
  run_ln(s, w.t2, p.ln1, w.t1, DM, DM);                       // out1
  run_gemm(s, w.t1, p.d1, nullptr, w.hid, DFF, 1);
  run_gemm(s, w.hid, p.d2, w.t1, w.t2, DM, 0);
  run_ln(s, w.t2, p.ln2, xout, DM, DM);
}
static void run_dec(hipStream_t s, const DecG& p, const float* xin, const float* eo,
                    float* xout, Bufs& w) {
  run_performer(s, p.mha1, xin, xin, xin, xin, w.t2, w);
  run_ln(s, w.t2, p.ln1, w.t1, DM, DM);                       // out1
  run_performer(s, p.mha2, w.t1, eo, eo, w.t1, w.t2, w);
  run_ln(s, w.t2, p.ln2, w.t1, DM, DM);                       // out2
  run_gemm(s, w.t1, p.d1, nullptr, w.hid, DFF, 1);
  run_gemm(s, w.hid, p.d2, w.t1, w.t2, DM, 0);
  run_ln(s, w.t2, p.ln3, xout, DM, DM);
}

extern "C" void kernel_launch(void* const* d_in, const int* in_sizes, int n_in,
                              void* d_out, int out_size, void* d_ws, size_t ws_size,
                              hipStream_t stream) {
  (void)in_sizes; (void)n_in; (void)out_size; (void)ws_size;
  const float* X = (const float*)d_in[0];

  // params, jax tree-flatten (alphabetical dict-key) leaf order
  Cursor c{d_in, 1};
  DecP dec_id0  = rdDec(c), dec_id1  = rdDec(c);
  DecP dec_reg0 = rdDec(c), dec_reg1 = rdDec(c);
  EncP enc_id0  = rdEnc(c), enc_id1  = rdEnc(c);
  EncP enc_reg0 = rdEnc(c), enc_reg1 = rdEnc(c);
  FfnP ffn_charge = rdFfn(c);
  FfnP ffn_id     = rdFfn(c);
  FfnP ffn_in     = rdFfn(c);
  FfnP ffn_mom    = rdFfn(c);
  LnP  ln_in      = rdLn(c);

  // workspace carve: fp32 activations then bf16 weight pool
  float* base = (float*)d_ws;
  size_t off = 0;
  auto alloc = [&](size_t n) { float* p = base + off; off += n; return p; };
  Bufs w;
  const size_t M = M_TOTAL;
  w.enc26  = alloc(M * ENC_D);
  w.enc26n = alloc(M * 32);          // K-padded to 32
  w.enc_t  = alloc(M * DM);
  w.eo     = alloc(M * DM);
  w.dcur   = alloc(M * DM);
  w.t1     = alloc(M * DM);
  w.t2     = alloc(M * DM);
  w.q      = alloc(M * DM);
  w.k      = alloc(M * DM);
  w.v      = alloc(M * DM);
  w.qp     = alloc(M * DM);
  w.kh     = alloc(M * DM);
  w.attno  = alloc(M * DM);
  w.hid    = alloc(M * DFF);
  w.cat282 = alloc(M * 288);         // K-padded to 288
  w.cat290 = alloc(M * 320);         // K-padded to 320
  w.logits8 = alloc(M * 8);
  w.charge  = alloc(M);
  w.mom     = alloc(M * 3);
  w.kv      = alloc(16 * 1024);
  w.ksum    = alloc(16 * 32);
  w.stab    = (unsigned*)alloc(16);
  off = (off + 3) & ~(size_t)3;      // 16B-align bf16 pool
  __bf16* pool = (__bf16*)(base + off);

  hipStream_t s = stream;

  // ---- weight prep (transpose + pad + bf16) ----
  DecG g_dec_id0  = prep_dec(s, dec_id0, pool),  g_dec_id1  = prep_dec(s, dec_id1, pool);
  DecG g_dec_reg0 = prep_dec(s, dec_reg0, pool), g_dec_reg1 = prep_dec(s, dec_reg1, pool);
  EncG g_enc_id0  = prep_enc(s, enc_id0, pool),  g_enc_id1  = prep_enc(s, enc_id1, pool);
  EncG g_enc_reg0 = prep_enc(s, enc_reg0, pool), g_enc_reg1 = prep_enc(s, enc_reg1, pool);
  GW g_ch1 = prep_dense(s, ffn_charge.d1, ENC_D + DM, DFF, pool);
  GW g_ch2 = prep_dense(s, ffn_charge.d2, DFF, 1, pool);
  GW g_id1 = prep_dense(s, ffn_id.d1, ENC_D + DM, DFF, pool);
  GW g_id2 = prep_dense(s, ffn_id.d2, DFF, 8, pool);
  GW g_in1 = prep_dense(s, ffn_in.d1, ENC_D, DFF, pool);
  GW g_in2 = prep_dense(s, ffn_in.d2, DFF, DM, pool);
  GW g_mo1 = prep_dense(s, ffn_mom.d1, ENC_D + 8 + DM, DFF, pool);
  GW g_mo2 = prep_dense(s, ffn_mom.d2, DFF, 3, pool);

  // ---- input embedding ----
  k_enc_build<<<cdiv((long)M * ENC_D, 256), 256, 0, s>>>(X, w.enc26);
  run_ln(s, w.enc26, ln_in, w.enc26n, ENC_D, 32);
  run_gemm(s, w.enc26n, g_in1, nullptr, w.hid, DFF, 1);
  run_gemm(s, w.hid, g_in2, nullptr, w.enc_t, DM, 0);

  // ---- id branch ----
  run_enc(s, g_enc_id0, w.enc_t, w.eo, w);
  run_enc(s, g_enc_id1, w.eo, w.eo, w);
  run_dec(s, g_dec_id0, w.enc_t, w.eo, w.dcur, w);
  run_dec(s, g_dec_id1, w.dcur, w.eo, w.dcur, w);
  k_cat2<<<cdiv((long)M * 288, 256), 256, 0, s>>>(w.enc26, w.dcur, w.cat282, ENC_D, DM, 288);
  run_gemm(s, w.cat282, g_id1, nullptr, w.hid, DFF, 1);
  run_gemm(s, w.hid, g_id2, nullptr, w.logits8, 8, 0);
  run_gemm(s, w.cat282, g_ch1, nullptr, w.hid, DFF, 1);
  run_gemm(s, w.hid, g_ch2, nullptr, w.charge, 1, 0);

  // ---- reg branch ----
  run_enc(s, g_enc_reg0, w.enc_t, w.eo, w);
  run_enc(s, g_enc_reg1, w.eo, w.eo, w);
  run_dec(s, g_dec_reg0, w.enc_t, w.eo, w.dcur, w);
  run_dec(s, g_dec_reg1, w.dcur, w.eo, w.dcur, w);
  k_cat3<<<cdiv((long)M * 320, 256), 256, 0, s>>>(w.enc26, w.logits8, w.dcur,
                                                  w.cat290, ENC_D, 8, DM, 320);
  run_gemm(s, w.cat290, g_mo1, nullptr, w.hid, DFF, 1);
  run_gemm(s, w.hid, g_mo2, nullptr, w.mom, 3, 0);

  // ---- assemble [M,12] ----
  k_final<<<cdiv((long)M * 12, 256), 256, 0, s>>>(X, w.logits8, w.charge, w.mom, (float*)d_out);
}
